// DmTranslateTrain_54820962566565
// MI455X (gfx1250) — compile-verified, matrix operand order checked by reading, not run on hardware
//
#include <hip/hip_runtime.h>
#include <hip/hip_bf16.h>

// Problem sizes (fixed by the reference)
#define B_   32
#define TS_  64
#define TD_  63
#define VT_  32000
#define E_   256
#define U_   1024

typedef __bf16 bf16_t;
typedef __attribute__((ext_vector_type(16))) __bf16 v16bf;
typedef __attribute__((ext_vector_type(8)))  float  v8f;

__device__ __forceinline__ float sigmoidf_(float x) { return 1.0f / (1.0f + __expf(-x)); }

// ---------------------------------------------------------------------------
// Weight convert + transpose: W[K][N] fp32  ->  out[n*ldo + k_off + k] bf16
// (stores B^T so WMMA B-fragments are contiguous 32B loads)
// ---------------------------------------------------------------------------
__global__ void wt_convert_kernel(const float* __restrict__ W, int K, int N,
                                  bf16_t* __restrict__ out, int ldo, int k_off) {
    long long idx = (long long)blockIdx.x * blockDim.x + threadIdx.x;
    long long total = (long long)K * N;
    if (idx >= total) return;
    int k = (int)(idx / N);
    int n = (int)(idx % N);
    out[(size_t)n * ldo + k_off + k] = (bf16_t)W[idx];
}

// ---------------------------------------------------------------------------
// WMMA GEMM: C[M,ldc] = A[M,K](bf16,row-major) * Bt[N,K](bf16 = B^T) + bias
// One wave computes a 32x16 strip (two 16x16 M-tiles sharing the B fragment).
// blockDim = 256 (8 waves); grid.x covers N/16 tiles, grid.y covers M/32.
// ---------------------------------------------------------------------------
__global__ void gemm_bf16_wmma(const bf16_t* __restrict__ A,
                               const bf16_t* __restrict__ Bt,
                               float* __restrict__ C,
                               const float* __restrict__ bias,
                               int M, int N, int K, int ldc) {
    int wave  = threadIdx.x >> 5;
    int lane  = threadIdx.x & 31;
    int ntile = blockIdx.x * (blockDim.x >> 5) + wave;
    if (ntile * 16 >= N) return;              // wave-uniform: EXEC stays all-1s
    int mbase = blockIdx.y * 32;

    int half = lane >> 4;                     // 0: K 0..15  1: K 16..31
    int l16  = lane & 15;

    const bf16_t* arow0 = A  + (size_t)(mbase + l16)      * K + half * 16;
    const bf16_t* arow1 = A  + (size_t)(mbase + 16 + l16) * K + half * 16;
    const bf16_t* brow  = Bt + (size_t)(ntile * 16 + l16) * K + half * 16;

    v8f acc0 = {};
    v8f acc1 = {};
    for (int k = 0; k < K; k += 32) {
        v16bf a0 = *(const v16bf*)(arow0 + k);
        v16bf a1 = *(const v16bf*)(arow1 + k);
        v16bf b  = *(const v16bf*)(brow + k);
        acc0 = __builtin_amdgcn_wmma_f32_16x16x32_bf16(false, a0, false, b,
                                                       (short)0, acc0, false, false);
        acc1 = __builtin_amdgcn_wmma_f32_16x16x32_bf16(false, a1, false, b,
                                                       (short)0, acc1, false, false);
    }

    int   col = ntile * 16 + l16;
    float bv  = bias ? bias[col] : 0.0f;
    // C/D layout: vgpr r, lanes 0-15 -> row r, lanes 16-31 -> row r+8, col = lane%16
    #pragma unroll
    for (int r = 0; r < 8; ++r) {
        int row0 = mbase + r + half * 8;
        int row1 = row0 + 16;
        C[(size_t)row0 * ldc + col] = acc0[r] + bv;
        C[(size_t)row1 * ldc + col] = acc1[r] + bv;
    }
}

// ---------------------------------------------------------------------------
// Build LSTM cell input (bf16): [ embed(tok) | seg1(optional) | seg2 ]
// encoder: ld = E+U   ( embed , h )        seg1 == nullptr
// decoder: ld = E+2U  ( embed , attn , h )
// ---------------------------------------------------------------------------
__global__ void build_cell_in_kernel(const int* __restrict__ tok, int tok_stride, int t,
                                     const float* __restrict__ emb,
                                     const bf16_t* __restrict__ seg1,
                                     const bf16_t* __restrict__ seg2,
                                     bf16_t* __restrict__ out, int ld) {
    int idx = blockIdx.x * blockDim.x + threadIdx.x;
    if (idx >= B_ * ld) return;
    int b = idx / ld;
    int j = idx % ld;
    if (j < E_) {
        int row = tok[b * tok_stride + t];
        out[idx] = (bf16_t)emb[(size_t)row * E_ + j];
    } else if (seg1 && j < E_ + U_) {
        out[idx] = seg1[b * U_ + (j - E_)];
    } else {
        int base = seg1 ? (E_ + U_) : E_;
        out[idx] = seg2[b * U_ + (j - base)];
    }
}

// ---------------------------------------------------------------------------
// LSTM gates (Keras order i,f,g,o); updates h/c fp32, h bf16; optionally
// writes encoder memory (fp32 [B,TS,U] and bf16 [B*TS,U]) at step t.
// ---------------------------------------------------------------------------
__global__ void lstm_gates_kernel(const float* __restrict__ z,
                                  float* __restrict__ h, float* __restrict__ c,
                                  bf16_t* __restrict__ h_bf,
                                  float* __restrict__ mem_f32,
                                  bf16_t* __restrict__ mem_bf, int t) {
    int idx = blockIdx.x * blockDim.x + threadIdx.x;
    if (idx >= B_ * U_) return;
    int b = idx / U_;
    int u = idx % U_;
    const float* zb = z + (size_t)b * 4 * U_;
    float gi = sigmoidf_(zb[u]);
    float gf = sigmoidf_(zb[U_ + u]);
    float gg = tanhf(zb[2 * U_ + u]);
    float go = sigmoidf_(zb[3 * U_ + u]);
    float cn = gf * c[idx] + gi * gg;
    float hn = go * tanhf(cn);
    c[idx]    = cn;
    h[idx]    = hn;
    h_bf[idx] = (bf16_t)hn;
    if (mem_f32) {
        mem_f32[(size_t)b * TS_ * U_ + (size_t)t * U_ + u] = hn;
        mem_bf[((size_t)b * TS_ + t) * U_ + u]             = (bf16_t)hn;
    }
}

// score[b,s] = dot(h[b,:], keys[b*TS+s,:]) ; one wave per (b,s)
__global__ void score_kernel(const float* __restrict__ h,
                             const float* __restrict__ keys,
                             float* __restrict__ score) {
    int wid  = (blockIdx.x * blockDim.x + threadIdx.x) >> 5;
    int lane = threadIdx.x & 31;
    if (wid >= B_ * TS_) return;
    int b = wid / TS_;
    const float* hb = h + (size_t)b * U_;
    const float* kb = keys + (size_t)wid * U_;
    float acc = 0.0f;
    for (int u = lane; u < U_; u += 32) acc += hb[u] * kb[u];
    #pragma unroll
    for (int off = 16; off > 0; off >>= 1) acc += __shfl_xor(acc, off, 32);
    if (lane == 0) score[wid] = acc;
}

// in-place softmax over TS=64 per batch row; block = 64 threads
__global__ void softmax_kernel(float* __restrict__ score) {
    __shared__ float red[TS_];
    int b = blockIdx.x;
    int s = threadIdx.x;
    float v = score[b * TS_ + s];
    red[s] = v;
    __syncthreads();
    for (int off = 32; off > 0; off >>= 1) {
        if (s < off) red[s] = fmaxf(red[s], red[s + off]);
        __syncthreads();
    }
    float m = red[0];
    __syncthreads();
    float e = __expf(v - m);
    red[s] = e;
    __syncthreads();
    for (int off = 32; off > 0; off >>= 1) {
        if (s < off) red[s] += red[s + off];
        __syncthreads();
    }
    score[b * TS_ + s] = e / red[0];
}

// context[b,u] = sum_s align[b,s] * memory[b,s,u]
__global__ void context_kernel(const float* __restrict__ align,
                               const float* __restrict__ mem,
                               float* __restrict__ ctx) {
    int b = blockIdx.x;
    for (int u = threadIdx.x; u < U_; u += blockDim.x) {
        float acc = 0.0f;
        for (int s = 0; s < TS_; ++s)
            acc += align[b * TS_ + s] * mem[(size_t)b * TS_ * U_ + (size_t)s * U_ + u];
        ctx[b * U_ + u] = acc;
    }
}

// attn_cat (bf16) = [ h_bf | bf16(context) ]
__global__ void attn_cat_kernel(const bf16_t* __restrict__ h_bf,
                                const float* __restrict__ ctx,
                                bf16_t* __restrict__ out) {
    int idx = blockIdx.x * blockDim.x + threadIdx.x;
    if (idx >= B_ * 2 * U_) return;
    int b = idx / (2 * U_);
    int j = idx % (2 * U_);
    out[idx] = (j < U_) ? h_bf[b * U_ + j] : (bf16_t)ctx[b * U_ + (j - U_)];
}

__global__ void f32_to_bf16_kernel(const float* __restrict__ in,
                                   bf16_t* __restrict__ out, int n) {
    int idx = blockIdx.x * blockDim.x + threadIdx.x;
    if (idx < n) out[idx] = (bf16_t)in[idx];
}

// ---------------------------------------------------------------------------
extern "C" void kernel_launch(void* const* d_in, const int* in_sizes, int n_in,
                              void* d_out, int out_size, void* d_ws, size_t ws_size,
                              hipStream_t stream) {
    (void)in_sizes; (void)n_in; (void)out_size; (void)ws_size;

    const int*   enc_tok = (const int*)  d_in[0];
    const int*   dec_tok = (const int*)  d_in[1];
    const float* enc_emb = (const float*)d_in[2];
    const float* dec_emb = (const float*)d_in[3];
    const float* Wx_e    = (const float*)d_in[4];
    const float* Wh_e    = (const float*)d_in[5];
    const float* b_e     = (const float*)d_in[6];
    const float* Wx_d    = (const float*)d_in[7];
    const float* Wh_d    = (const float*)d_in[8];
    const float* b_d     = (const float*)d_in[9];
    const float* Wm      = (const float*)d_in[10];
    const float* Wa      = (const float*)d_in[11];
    const float* Wf      = (const float*)d_in[12];
    const float* bfv     = (const float*)d_in[13];
    float* out = (float*)d_out;

    // ---- workspace bump allocator (256B aligned) ----
    char* ws = (char*)d_ws;
    size_t off = 0;
    auto alloc = [&](size_t bytes) -> char* {
        char* p = ws + off;
        off = (off + bytes + 255) & ~(size_t)255;
        return p;
    };
    bf16_t* WeT    = (bf16_t*)alloc((size_t)4 * U_ * (E_ + U_)     * 2); // [4096][1280]
    bf16_t* WdT    = (bf16_t*)alloc((size_t)4 * U_ * (E_ + 2 * U_) * 2); // [4096][2304]
    bf16_t* WmT    = (bf16_t*)alloc((size_t)U_ * U_ * 2);
    bf16_t* WaT    = (bf16_t*)alloc((size_t)U_ * 2 * U_ * 2);
    bf16_t* WfT    = (bf16_t*)alloc((size_t)VT_ * U_ * 2);               // 64 MB (L2-resident)
    float*  memF   = (float*) alloc((size_t)B_ * TS_ * U_ * 4);
    bf16_t* memBF  = (bf16_t*)alloc((size_t)B_ * TS_ * U_ * 2);
    float*  keys   = (float*) alloc((size_t)B_ * TS_ * U_ * 4);
    float*  zbuf   = (float*) alloc((size_t)B_ * 4 * U_ * 4);
    float*  h      = (float*) alloc((size_t)B_ * U_ * 4);
    float*  c      = (float*) alloc((size_t)B_ * U_ * 4);
    bf16_t* hBF    = (bf16_t*)alloc((size_t)B_ * U_ * 2);
    float*  attn   = (float*) alloc((size_t)B_ * U_ * 4);
    bf16_t* attnBF = (bf16_t*)alloc((size_t)B_ * U_ * 2);
    float*  ctx    = (float*) alloc((size_t)B_ * U_ * 4);
    bf16_t* cellin = (bf16_t*)alloc((size_t)B_ * (E_ + 2 * U_) * 2);
    float*  score  = (float*) alloc((size_t)B_ * TS_ * 4);
    bf16_t* acat   = (bf16_t*)alloc((size_t)B_ * 2 * U_ * 2);

    // ---- weight conversion (fp32 -> transposed bf16), fused Wx||Wh blocks ----
    auto conv = [&](const float* W, int K, int N, bf16_t* o, int ldo, int k_off) {
        long long total = (long long)K * N;
        int blocks = (int)((total + 255) / 256);
        wt_convert_kernel<<<blocks, 256, 0, stream>>>(W, K, N, o, ldo, k_off);
    };
    conv(Wx_e, E_,        4 * U_, WeT, E_ + U_,     0);
    conv(Wh_e, U_,        4 * U_, WeT, E_ + U_,     E_);
    conv(Wx_d, E_ + U_,   4 * U_, WdT, E_ + 2 * U_, 0);
    conv(Wh_d, U_,        4 * U_, WdT, E_ + 2 * U_, E_ + U_);
    conv(Wm,   U_,        U_,     WmT, U_,          0);
    conv(Wa,   2 * U_,    U_,     WaT, 2 * U_,      0);
    conv(Wf,   U_,        VT_,    WfT, U_,          0);

    // ---- initial state: h0 = c0 = 0, attn0 = 0 ----
    hipMemsetAsync(h,      0, (size_t)B_ * U_ * 4, stream);
    hipMemsetAsync(c,      0, (size_t)B_ * U_ * 4, stream);
    hipMemsetAsync(hBF,    0, (size_t)B_ * U_ * 2, stream);
    hipMemsetAsync(attnBF, 0, (size_t)B_ * U_ * 2, stream);

    auto gemm = [&](const bf16_t* A, const bf16_t* Bt, float* C, const float* bias,
                    int M, int N, int K, int ldc) {
        dim3 grid((N / 16 + 7) / 8, M / 32);
        gemm_bf16_wmma<<<grid, 256, 0, stream>>>(A, Bt, C, bias, M, N, K, ldc);
    };

    // ---- encoder: 64 sequential LSTM steps, fused [x|h] GEMM ----
    for (int t = 0; t < TS_; ++t) {
        build_cell_in_kernel<<<(B_ * (E_ + U_) + 255) / 256, 256, 0, stream>>>(
            enc_tok, TS_, t, enc_emb, nullptr, hBF, cellin, E_ + U_);
        gemm(cellin, WeT, zbuf, b_e, B_, 4 * U_, E_ + U_, 4 * U_);
        lstm_gates_kernel<<<(B_ * U_ + 255) / 256, 256, 0, stream>>>(
            zbuf, h, c, hBF, memF, memBF, t);
    }

    // ---- attention keys: one big GEMM over all memory rows ----
    gemm(memBF, WmT, keys, nullptr, B_ * TS_, U_, U_, U_);

    // ---- decoder: 63 sequential attention-LSTM steps ----
    for (int t = 0; t < TD_; ++t) {
        build_cell_in_kernel<<<(B_ * (E_ + 2 * U_) + 255) / 256, 256, 0, stream>>>(
            dec_tok, TD_, t, dec_emb, attnBF, hBF, cellin, E_ + 2 * U_);
        gemm(cellin, WdT, zbuf, b_d, B_, 4 * U_, E_ + 2 * U_, 4 * U_);
        lstm_gates_kernel<<<(B_ * U_ + 255) / 256, 256, 0, stream>>>(
            zbuf, h, c, hBF, nullptr, nullptr, 0);
        score_kernel<<<(B_ * TS_ * 32 + 255) / 256, 256, 0, stream>>>(h, keys, score);
        softmax_kernel<<<B_, TS_, 0, stream>>>(score);
        context_kernel<<<B_, 256, 0, stream>>>(score, memF, ctx);
        attn_cat_kernel<<<(B_ * 2 * U_ + 255) / 256, 256, 0, stream>>>(hBF, ctx, acat);
        gemm(acat, WaT, attn, nullptr, B_, U_, 2 * U_, U_);
        f32_to_bf16_kernel<<<(B_ * U_ + 255) / 256, 256, 0, stream>>>(attn, attnBF, B_ * U_);
        // logits written straight into d_out[b, t, :] via ldc = TD*VT
        gemm(attnBF, WfT, out + (size_t)t * VT_, bfv, B_, VT_, U_, TD_ * VT_);
    }
}
